// CNNEncoder_15934328669000
// MI455X (gfx1250) — matrix operand order repaired and checked by reading, not compiled, not run on hardware
//
#include <hip/hip_runtime.h>

// ---------------------------------------------------------------------------
// Problem constants (match reference)
// ---------------------------------------------------------------------------
#define BB   32
#define NN   20
#define TT   100
#define DD   4
#define HH   128
#define EE   (NN * (NN - 1))       // 380
#define BE   (BB * EE)             // 12160
#define L1   96                    // conv1 valid length
#define L1P  48                    // after maxpool k=2
#define L2   44                    // conv2 valid length
#define BNODE (BB * NN)            // 640
#define NOUT 16

typedef _Float16 f16;
typedef __attribute__((ext_vector_type(16))) _Float16 v16h;
typedef __attribute__((ext_vector_type(8)))  float    v8f;

// ---------------------------------------------------------------------------
// WMMA tile helpers (force-inlined; NT is compile-time so guards fold away)
// ---------------------------------------------------------------------------
template <int NT>
__device__ __forceinline__ void wmma_tile(int nt, int lr, int hi,
                                          const v16h& a, const f16 BsT[128][40],
                                          v8f& acc)
{
    if (nt >= NT) return;           // compile-time constant fold
    union { uint4 u[2]; v16h v; } ub;
    const int c = nt * 16 + lr;
    ub.u[0] = *(const uint4*)&BsT[c][hi * 8];
    ub.u[1] = *(const uint4*)&BsT[c][hi * 8 + 16];
    acc = __builtin_amdgcn_wmma_f32_16x16x32_f16(
        false, a, false, ub.v, (short)0, acc, false, false);
}

template <int NT>
__device__ __forceinline__ void store_tile(int nt, int m0, int n0,
                                           int wave, int lr, int hi,
                                           const v8f& acc,
                                           const float* __restrict__ bias,
                                           void* __restrict__ Cout,
                                           int M, int Ncols, int act, int outF32)
{
    if (nt >= NT) return;           // compile-time constant fold
    #pragma unroll
    for (int g = 0; g < 8; ++g) {
        const int r = m0 + wave * 16 + hi * 8 + g;
        const int c = n0 + nt * 16 + lr;
        if (r < M) {
            float v = acc[g];
            if (bias) v += bias[c];
            if (act == 1) v = fmaxf(v, 0.0f);
            else if (act == 2) v = (v > 0.0f) ? v : expm1f(v);
            if (outF32) ((float*)Cout)[(long)r * Ncols + c] = v;
            else        ((f16*)Cout)[(long)r * Ncols + c] = (f16)v;
        }
    }
}

// ---------------------------------------------------------------------------
// Generic f16 WMMA GEMM:  C[M][Ncols] = act(A[M][K] * B[K][Ncols] + bias)
//   NT (template): number of 16-wide column tiles this launch covers (1..8)
//   n0base: column offset of this launch (plus blockIdx.y*128)
//   aMode 0: A row r at A + r*K        (plain row-major)
//   aMode 1: A row r at A + (r/44)*48*128 + (r%44)*128   (conv2 im2col view)
//   act: 0 none, 1 relu, 2 elu ; outF32: store float else f16
// Block: 256 threads = 8 waves (wave32). Each wave owns 16 M-rows and all NT
// column tiles. K stepped by 32 using v_wmma_f32_16x16x32_f16.
// ---------------------------------------------------------------------------
template <int NT>
__global__ __launch_bounds__(256)
void gemm_f16_wmma(const f16* __restrict__ A, const f16* __restrict__ Bm,
                   const float* __restrict__ bias, void* __restrict__ Cout,
                   int M, int K, int Ncols, int n0base, int act, int aMode, int outF32)
{
    __shared__ f16 As[128][40];   // 128 rows x 32 halfs (+8 pad)
    __shared__ f16 BsT[128][40];  // transposed: [col][k] 128 x 32 (+8 pad)

    const int tid  = threadIdx.x;
    const int wave = tid >> 5;
    const int lane = tid & 31;
    const int lr   = lane & 15;
    const int hi   = lane >> 4;
    const int m0   = blockIdx.x * 128;
    const int n0   = n0base + blockIdx.y * 128;

    v8f acc0, acc1, acc2, acc3, acc4, acc5, acc6, acc7;
    #pragma unroll
    for (int j = 0; j < 8; ++j) {
        acc0[j] = 0.0f; acc1[j] = 0.0f; acc2[j] = 0.0f; acc3[j] = 0.0f;
        acc4[j] = 0.0f; acc5[j] = 0.0f; acc6[j] = 0.0f; acc7[j] = 0.0f;
    }

    // A staging role: each thread copies 16 halfs of one row
    const int arow = tid >> 1;        // 0..127
    const int aseg = (tid & 1) * 16;  // 0 or 16
    long aoff = -1;
    {
        const int gr = m0 + arow;
        if (gr < M) {
            if (aMode == 1) aoff = (long)(gr / 44) * (48 * 128) + (long)(gr % 44) * 128;
            else            aoff = (long)gr * K;
        }
    }
    // B staging role: thread fetches 16 halfs of row (k0+bk), cols [n0+bc, +16)
    const int bk = tid >> 3;          // 0..31
    const int bc = (tid & 7) * 16;    // 0..112
    const bool bIn = (n0 + bc + 16) <= Ncols;

    for (int k0 = 0; k0 < K; k0 += 32) {
        // ---- fetch from global into registers ----
        uint4 av0 = make_uint4(0, 0, 0, 0), av1 = av0;
        if (aoff >= 0) {
            const uint4* p = (const uint4*)(A + aoff + k0 + aseg);
            av0 = p[0]; av1 = p[1];
            if (k0 + 32 < K) __builtin_prefetch(A + aoff + k0 + 32, 0, 0);
        }
        f16 btmp[16];
        if (bIn) {
            const uint4* p = (const uint4*)(Bm + (long)(k0 + bk) * Ncols + n0 + bc);
            union { uint4 u; f16 h[8]; } u0, u1;
            u0.u = p[0]; u1.u = p[1];
            #pragma unroll
            for (int x = 0; x < 8; ++x) { btmp[x] = u0.h[x]; btmp[8 + x] = u1.h[x]; }
        } else {
            #pragma unroll
            for (int x = 0; x < 16; ++x) btmp[x] = (f16)0.0f;
        }

        __syncthreads();   // previous iteration's LDS reads done
        *(uint4*)&As[arow][aseg]     = av0;
        *(uint4*)&As[arow][aseg + 8] = av1;
        #pragma unroll
        for (int x = 0; x < 16; ++x) BsT[bc + x][bk] = btmp[x];
        __syncthreads();   // staging complete

        // ---- A fragment (16x32 f16): lanes 0-15 rows; K 0..7/16..23, hi half +8 ----
        v16h a;
        {
            union { uint4 u[2]; v16h v; } ua;
            const int r = wave * 16 + lr;
            ua.u[0] = *(const uint4*)&As[r][hi * 8];
            ua.u[1] = *(const uint4*)&As[r][hi * 8 + 16];
            a = ua.v;
        }
        // ---- NT n-tiles, named accumulators, zero per-tile control flow ----
        wmma_tile<NT>(0, lr, hi, a, BsT, acc0);
        wmma_tile<NT>(1, lr, hi, a, BsT, acc1);
        wmma_tile<NT>(2, lr, hi, a, BsT, acc2);
        wmma_tile<NT>(3, lr, hi, a, BsT, acc3);
        wmma_tile<NT>(4, lr, hi, a, BsT, acc4);
        wmma_tile<NT>(5, lr, hi, a, BsT, acc5);
        wmma_tile<NT>(6, lr, hi, a, BsT, acc6);
        wmma_tile<NT>(7, lr, hi, a, BsT, acc7);
    }

    store_tile<NT>(0, m0, n0, wave, lr, hi, acc0, bias, Cout, M, Ncols, act, outF32);
    store_tile<NT>(1, m0, n0, wave, lr, hi, acc1, bias, Cout, M, Ncols, act, outF32);
    store_tile<NT>(2, m0, n0, wave, lr, hi, acc2, bias, Cout, M, Ncols, act, outF32);
    store_tile<NT>(3, m0, n0, wave, lr, hi, acc3, bias, Cout, M, Ncols, act, outF32);
    store_tile<NT>(4, m0, n0, wave, lr, hi, acc4, bias, Cout, M, Ncols, act, outF32);
    store_tile<NT>(5, m0, n0, wave, lr, hi, acc5, bias, Cout, M, Ncols, act, outF32);
    store_tile<NT>(6, m0, n0, wave, lr, hi, acc6, bias, Cout, M, Ncols, act, outF32);
    store_tile<NT>(7, m0, n0, wave, lr, hi, acc7, bias, Cout, M, Ncols, act, outF32);
}

// ---------------------------------------------------------------------------
// Weight prep kernels
// ---------------------------------------------------------------------------
__global__ void prep_w1_kernel(const float* __restrict__ w, f16* __restrict__ w1f)
{
    // w1f[kk][c], kk = k*4+ci (matches contiguous inputs span), c<128 send / c>=128 recv
    int idx = blockIdx.x * blockDim.x + threadIdx.x;
    if (idx >= 32 * 256) return;
    int kk = idx / 256, c = idx % 256;
    f16 v = (f16)0.0f;
    if (kk < 20) {
        int k = kk / 4, ci = kk % 4;
        int cin = (c < 128) ? ci : ci + 4;
        int co  = c & 127;
        v = (f16)w[(co * 8 + cin) * 5 + k];
    }
    w1f[idx] = v;
}

__global__ void prep_w2_kernel(const float* __restrict__ w, f16* __restrict__ w2f)
{
    // w2f[kt*128+ci][co] = conv2_w[co][ci][kt]
    int idx = blockIdx.x * blockDim.x + threadIdx.x;
    if (idx >= 640 * 128) return;
    int kk = idx / 128, co = idx % 128;
    int kt = kk / 128, ci = kk % 128;
    w2f[idx] = (f16)w[(co * 128 + ci) * 5 + kt];
}

__global__ void transpose_w_kernel(const float* __restrict__ src, f16* __restrict__ dst,
                                   int O, int I)
{
    // src [O][I] -> dst [I][O]
    long idx = (long)blockIdx.x * blockDim.x + threadIdx.x;
    if (idx >= (long)O * I) return;
    int o = (int)(idx / I), i = (int)(idx % I);
    dst[(long)i * O + o] = (f16)src[idx];
}

__global__ void im2col1_kernel(const float* __restrict__ inp, f16* __restrict__ A1)
{
    // Row r = bn*96 + t : contiguous 20 floats inputs[bn][t..t+4][0..3], pad to 32
    int r = blockIdx.x * blockDim.x + threadIdx.x;
    if (r >= BNODE * L1) return;
    int bn = r / L1, t = r % L1;
    const float* src = inp + (long)bn * (TT * DD) + t * DD;
    f16* dst = A1 + (long)r * 32;
    #pragma unroll
    for (int p = 0; p < 20; ++p) dst[p] = (f16)src[p];
    #pragma unroll
    for (int p = 20; p < 32; ++p) dst[p] = (f16)0.0f;
}

// ---------------------------------------------------------------------------
// BN statistics / finalize / affine
// ---------------------------------------------------------------------------
__global__ void stats1_kernel(const f16* __restrict__ y1, const float* __restrict__ cb,
                              float* __restrict__ out)
{
    // per-channel sum/sumsq of relu(ys + yr + bias) over (b,e,t)
    const int c = threadIdx.x;   // 0..127
    const float b = cb[c];
    float s = 0.0f, ss = 0.0f;
    const long total = (long)BE * L1;
    for (long idx = blockIdx.x; idx < total; idx += gridDim.x) {
        long be = idx / L1; int t = (int)(idx % L1);
        int bb = (int)(be / EE); int e = (int)(be % EE);
        int i = e / (NN - 1); int off = e % (NN - 1); int j = off + (off >= i ? 1 : 0);
        long rs = ((long)(bb * NN + j) * L1 + t);
        long rr = ((long)(bb * NN + i) * L1 + t);
        float v = (float)y1[rs * 256 + c] + (float)y1[rr * 256 + 128 + c] + b;
        v = fmaxf(v, 0.0f);
        s += v; ss += v * v;
    }
    atomicAdd(&out[c], s);
    atomicAdd(&out[128 + c], ss);
}

__global__ void stats_rows_kernel(const f16* __restrict__ X, long rows,
                                  float* __restrict__ out)
{
    const int c = threadIdx.x;   // 0..127
    float s = 0.0f, ss = 0.0f;
    for (long r = blockIdx.x; r < rows; r += gridDim.x) {
        float v = (float)X[r * 128 + c];
        s += v; ss += v * v;
    }
    atomicAdd(&out[c], s);
    atomicAdd(&out[128 + c], ss);
}

__global__ void bn_finalize_kernel(const float* __restrict__ stats, float cnt,
                                   const float* __restrict__ g, const float* __restrict__ bsh,
                                   float* __restrict__ ac)
{
    int c = threadIdx.x;
    if (c >= 128) return;
    float m   = stats[c] / cnt;
    float var = stats[128 + c] / cnt - m * m;
    float a   = g[c] * rsqrtf(var + 1e-5f);
    ac[c] = a;
    ac[128 + c] = bsh[c] - m * a;
}

__global__ void affine_kernel(const f16* __restrict__ X, const float* __restrict__ ac,
                              f16* __restrict__ Y, long total)
{
    long idx = (long)blockIdx.x * blockDim.x + threadIdx.x;
    if (idx >= total) return;
    int c = (int)(idx & 127);
    Y[idx] = (f16)((float)X[idx] * ac[c] + ac[128 + c]);
}

// ---------------------------------------------------------------------------
// conv1 edge-combine + BN1 + maxpool(k2) -> pooled [BE][48][128] f16
// ---------------------------------------------------------------------------
__global__ void pool_kernel(const f16* __restrict__ y1, const float* __restrict__ cb,
                            const float* __restrict__ ac, f16* __restrict__ pooled)
{
    const int c = threadIdx.x;           // channel
    const long idx = blockIdx.x;         // (b*E+e)*48 + tp
    long be = idx / L1P; int tp = (int)(idx % L1P);
    int bb = (int)(be / EE); int e = (int)(be % EE);
    int i = e / (NN - 1); int off = e % (NN - 1); int j = off + (off >= i ? 1 : 0);
    const float a = ac[c], cc = ac[128 + c], bv = cb[c];
    float best = -1e30f;
    #pragma unroll
    for (int dt = 0; dt < 2; ++dt) {
        int t = 2 * tp + dt;
        long rs = ((long)(bb * NN + j) * L1 + t);
        long rr = ((long)(bb * NN + i) * L1 + t);
        float v = (float)y1[rs * 256 + c] + (float)y1[rr * 256 + 128 + c] + bv;
        v = fmaxf(v, 0.0f) * a + cc;
        best = fmaxf(best, v);
    }
    pooled[idx * 128 + c] = (f16)best;
}

// ---------------------------------------------------------------------------
// Fold BN2 into pred/att head weights: headW f16 [128(c)][144(o)], headB f32[144]
// ---------------------------------------------------------------------------
__global__ void head_fold_kernel(const float* __restrict__ ac2,
                                 const float* __restrict__ pw, const float* __restrict__ pb,
                                 const float* __restrict__ aw, const float* __restrict__ ab,
                                 f16* __restrict__ headW, float* __restrict__ headB)
{
    int o = threadIdx.x;   // 0..127
    for (int c = 0; c < 128; ++c)
        headW[c * 144 + o] = (f16)(pw[o * 128 + c] * ac2[c]);
    float hb = pb[o];
    for (int c = 0; c < 128; ++c) hb += ac2[128 + c] * pw[o * 128 + c];
    headB[o] = hb;
    if (o == 0) {
        for (int c = 0; c < 128; ++c) headW[c * 144 + 128] = (f16)(aw[c] * ac2[c]);
        for (int oo = 129; oo < 144; ++oo)
            for (int c = 0; c < 128; ++c) headW[c * 144 + oo] = (f16)0.0f;
        float abv = ab[0];
        for (int c = 0; c < 128; ++c) abv += ac2[128 + c] * aw[c];
        headB[128] = abv;
        for (int oo = 129; oo < 144; ++oo) headB[oo] = 0.0f;
    }
}

// softmax over t of att column, attention-weighted mean of pred -> x1 [BE][128]
__global__ void head_reduce_kernel(const f16* __restrict__ pa, f16* __restrict__ x1)
{
    const long be = blockIdx.x;
    const int o = threadIdx.x;   // 0..127
    __shared__ float att[L2];
    if (o < L2) att[o] = (float)pa[(be * L2 + o) * 144 + 128];
    __syncthreads();
    float mx = -1e30f;
    for (int t = 0; t < L2; ++t) mx = fmaxf(mx, att[t]);
    float sm = 0.0f;
    for (int t = 0; t < L2; ++t) sm += __expf(att[t] - mx);
    float acc = 0.0f;
    for (int t = 0; t < L2; ++t) {
        float w = __expf(att[t] - mx) / sm;
        acc += w * (float)pa[(be * L2 + t) * 144 + o];
    }
    x1[be * 128 + o] = (f16)(acc * (1.0f / (float)L2));
}

// ---------------------------------------------------------------------------
// edge2node (contiguous receiver blocks) and concat builder
// ---------------------------------------------------------------------------
__global__ void edge2node_kernel(const f16* __restrict__ xe, f16* __restrict__ xn)
{
    const int c  = threadIdx.x;
    const int bn = blockIdx.x;           // b*N + i
    const int bb = bn / NN, i = bn % NN;
    float s = 0.0f;
    const long base = (long)bb * EE + (long)i * (NN - 1);
    for (int k = 0; k < NN - 1; ++k) s += (float)xe[(base + k) * 128 + c];
    xn[(long)bn * 128 + c] = (f16)(s / (float)NN);
}

__global__ void concat_kernel(const f16* __restrict__ m2b, const float* __restrict__ ac4,
                              const f16* __restrict__ xsk, f16* __restrict__ cat)
{
    const int c = threadIdx.x;
    const long row = blockIdx.x;         // b*E + e
    int bb = (int)(row / EE); int e = (int)(row % EE);
    int i = e / (NN - 1); int off = e % (NN - 1); int j = off + (off >= i ? 1 : 0);
    const float a = ac4[c], cc = ac4[128 + c];
    float es = (float)m2b[((long)(bb * NN + j)) * 128 + c] * a + cc;
    float er = (float)m2b[((long)(bb * NN + i)) * 128 + c] * a + cc;
    cat[row * 384 + c]       = (f16)es;
    cat[row * 384 + 128 + c] = (f16)er;
    cat[row * 384 + 256 + c] = xsk[row * 128 + c];
}

// fold mlp3 BN into fc_out: fcWT f16 [128][16], fcB f32[16]
__global__ void fc_fold_kernel(const float* __restrict__ ac5, const float* __restrict__ fw,
                               const float* __restrict__ fb, f16* __restrict__ fcWT,
                               float* __restrict__ fcB)
{
    int o = threadIdx.x;
    if (o >= NOUT) return;
    float b = fb[o];
    for (int c = 0; c < 128; ++c) {
        fcWT[c * NOUT + o] = (f16)(fw[o * 128 + c] * ac5[c]);
        b += ac5[128 + c] * fw[o * 128 + c];
    }
    fcB[o] = b;
}

// ---------------------------------------------------------------------------
// Launcher
// ---------------------------------------------------------------------------
extern "C" void kernel_launch(void* const* d_in, const int* in_sizes, int n_in,
                              void* d_out, int out_size, void* d_ws, size_t ws_size,
                              hipStream_t stream)
{
    (void)in_sizes; (void)n_in; (void)out_size; (void)ws_size;
    const float* inputs  = (const float*)d_in[0];
    const float* conv1_w = (const float*)d_in[3];
    const float* conv1_b = (const float*)d_in[4];
    const float* bn1_g   = (const float*)d_in[5];
    const float* bn1_b   = (const float*)d_in[6];
    const float* conv2_w = (const float*)d_in[7];
    const float* conv2_b = (const float*)d_in[8];
    const float* bn2_g   = (const float*)d_in[9];
    const float* bn2_b   = (const float*)d_in[10];
    const float* pred_w  = (const float*)d_in[11];
    const float* pred_b  = (const float*)d_in[12];
    const float* att_w   = (const float*)d_in[13];
    const float* att_b   = (const float*)d_in[14];
    const float* m1w1 = (const float*)d_in[15]; const float* m1b1 = (const float*)d_in[16];
    const float* m1w2 = (const float*)d_in[17]; const float* m1b2 = (const float*)d_in[18];
    const float* m1g  = (const float*)d_in[19]; const float* m1bb = (const float*)d_in[20];
    const float* m2w1 = (const float*)d_in[21]; const float* m2b1 = (const float*)d_in[22];
    const float* m2w2 = (const float*)d_in[23]; const float* m2b2 = (const float*)d_in[24];
    const float* m2g  = (const float*)d_in[25]; const float* m2bb = (const float*)d_in[26];
    const float* m3w1 = (const float*)d_in[27]; const float* m3b1 = (const float*)d_in[28];
    const float* m3w2 = (const float*)d_in[29]; const float* m3b2 = (const float*)d_in[30];
    const float* m3g  = (const float*)d_in[31]; const float* m3bb = (const float*)d_in[32];
    const float* fcw  = (const float*)d_in[33]; const float* fcb  = (const float*)d_in[34];

    // ---- workspace bump allocator ----
    char* p = (char*)d_ws;
    auto take = [&](size_t bytes) -> char* {
        char* r = p;
        p += (bytes + 255) & ~(size_t)255;
        return r;
    };

    f16* A1   = (f16*)take((size_t)BNODE * L1 * 32 * 2);       // 3.9 MB
    f16* w1f  = (f16*)take(32 * 256 * 2);
    // Region shared between {y1 + pooled} (early) and predatt (late)
    const size_t y1Bytes = (size_t)BNODE * L1 * 256 * 2;       // 31.5 MB
    const size_t plBytes = (size_t)BE * L1P * 128 * 2;         // 149.4 MB
    const size_t paBytes = (size_t)BE * L2 * 144 * 2;          // 154.1 MB
    size_t regionBytes = ((y1Bytes + 255) & ~(size_t)255) + plBytes;
    if (paBytes > regionBytes) regionBytes = paBytes;
    char* regY   = take(regionBytes);
    f16* y1      = (f16*)regY;
    f16* pooled  = (f16*)(regY + ((y1Bytes + 255) & ~(size_t)255));
    f16* predatt = (f16*)regY;                                  // aliases y1/pooled (dead)
    f16* w2f  = (f16*)take(640 * 128 * 2);
    f16* h2   = (f16*)take((size_t)BE * L2 * 128 * 2);          // 137 MB
    f16* headW = (f16*)take(128 * 144 * 2);
    float* headB = (float*)take(144 * 4);
    f16* x1   = (f16*)take((size_t)BE * 128 * 2);
    f16* m1a  = (f16*)take((size_t)BE * 128 * 2);
    f16* m1b  = (f16*)take((size_t)BE * 128 * 2);
    f16* xsk  = (f16*)take((size_t)BE * 128 * 2);
    f16* xn   = (f16*)take((size_t)BNODE * 128 * 2);
    f16* m2a  = (f16*)take((size_t)BNODE * 128 * 2);
    f16* m2b  = (f16*)take((size_t)BNODE * 128 * 2);
    f16* cat  = (f16*)take((size_t)BE * 384 * 2);
    f16* m3a  = (f16*)take((size_t)BE * 128 * 2);
    f16* m3b  = (f16*)take((size_t)BE * 128 * 2);
    f16* m1w1T = (f16*)take(128 * 128 * 2);
    f16* m1w2T = (f16*)take(128 * 128 * 2);
    f16* m2w1T = (f16*)take(128 * 128 * 2);
    f16* m2w2T = (f16*)take(128 * 128 * 2);
    f16* m3w1T = (f16*)take(384 * 128 * 2);
    f16* m3w2T = (f16*)take(128 * 128 * 2);
    f16* fcWT  = (f16*)take(128 * NOUT * 2);
    float* fcB = (float*)take(NOUT * 4);
    float* statsAll = (float*)take(5 * 256 * 4);
    float* acAll    = (float*)take(5 * 256 * 4);

    hipMemsetAsync(statsAll, 0, 5 * 256 * 4, stream);

    // ---- weight prep ----
    prep_w1_kernel<<<(32 * 256 + 255) / 256, 256, 0, stream>>>(conv1_w, w1f);
    prep_w2_kernel<<<(640 * 128 + 255) / 256, 256, 0, stream>>>(conv2_w, w2f);
    transpose_w_kernel<<<(128 * 128 + 255) / 256, 256, 0, stream>>>(m1w1, m1w1T, 128, 128);
    transpose_w_kernel<<<(128 * 128 + 255) / 256, 256, 0, stream>>>(m1w2, m1w2T, 128, 128);
    transpose_w_kernel<<<(128 * 128 + 255) / 256, 256, 0, stream>>>(m2w1, m2w1T, 128, 128);
    transpose_w_kernel<<<(128 * 128 + 255) / 256, 256, 0, stream>>>(m2w2, m2w2T, 128, 128);
    transpose_w_kernel<<<(128 * 384 + 255) / 256, 256, 0, stream>>>(m3w1, m3w1T, 128, 384);
    transpose_w_kernel<<<(128 * 128 + 255) / 256, 256, 0, stream>>>(m3w2, m3w2T, 128, 128);

    // ---- conv1 (per-node factorized): y1[bn*96+t][0:128]=send half, [128:256]=recv half
    im2col1_kernel<<<(BNODE * L1 + 255) / 256, 256, 0, stream>>>(inputs, A1);
    gemm_f16_wmma<8><<<dim3((BNODE * L1) / 128, 2), 256, 0, stream>>>(
        A1, w1f, nullptr, y1, BNODE * L1, 32, 256, 0, 0, 0, 0);

    // ---- BN1 stats over edge-expanded relu(conv1), then pool ----
    stats1_kernel<<<2048, 128, 0, stream>>>(y1, conv1_b, statsAll + 0);
    bn_finalize_kernel<<<1, 128, 0, stream>>>(statsAll + 0, (float)((long)BE * L1),
                                              bn1_g, bn1_b, acAll + 0);
    pool_kernel<<<(long)BE * L1P, 128, 0, stream>>>(y1, conv1_b, acAll + 0, pooled);

    // ---- conv2 as GEMM (M=BE*44, K=640, N=128), relu fused ----
    gemm_f16_wmma<8><<<dim3((BE * L2) / 128, 1), 256, 0, stream>>>(
        pooled, w2f, conv2_b, h2, BE * L2, 640, 128, 0, 1, 1, 0);

    // ---- BN2 stats + fold into pred/att head ----
    stats_rows_kernel<<<1024, 128, 0, stream>>>(h2, (long)BE * L2, statsAll + 256);
    bn_finalize_kernel<<<1, 128, 0, stream>>>(statsAll + 256, (float)((long)BE * L2),
                                              bn2_g, bn2_b, acAll + 256);
    head_fold_kernel<<<1, 128, 0, stream>>>(acAll + 256, pred_w, pred_b, att_w, att_b,
                                            headW, headB);

    // ---- head GEMM (M=BE*44, K=128, N=144): cols 0-127 (NT=8) + col 128-143 (NT=1) ----
    gemm_f16_wmma<8><<<dim3((BE * L2) / 128, 1), 256, 0, stream>>>(
        h2, headW, headB, predatt, BE * L2, 128, 144, 0, 0, 0, 0);
    gemm_f16_wmma<1><<<dim3((BE * L2) / 128, 1), 256, 0, stream>>>(
        h2, headW, headB, predatt, BE * L2, 128, 144, 128, 0, 0, 0);
    head_reduce_kernel<<<BE, 128, 0, stream>>>(predatt, x1);

    // ---- mlp1 + BN ----
    gemm_f16_wmma<8><<<dim3(BE / 128, 1), 256, 0, stream>>>(x1, m1w1T, m1b1, m1a, BE, 128, 128, 0, 2, 0, 0);
    gemm_f16_wmma<8><<<dim3(BE / 128, 1), 256, 0, stream>>>(m1a, m1w2T, m1b2, m1b, BE, 128, 128, 0, 2, 0, 0);
    stats_rows_kernel<<<1024, 128, 0, stream>>>(m1b, (long)BE, statsAll + 512);
    bn_finalize_kernel<<<1, 128, 0, stream>>>(statsAll + 512, (float)BE, m1g, m1bb, acAll + 512);
    affine_kernel<<<((long)BE * 128 + 255) / 256, 256, 0, stream>>>(m1b, acAll + 512, xsk, (long)BE * 128);

    // ---- edge2node, mlp2 + BN ----
    edge2node_kernel<<<BNODE, 128, 0, stream>>>(xsk, xn);
    gemm_f16_wmma<8><<<dim3(BNODE / 128, 1), 256, 0, stream>>>(xn, m2w1T, m2b1, m2a, BNODE, 128, 128, 0, 2, 0, 0);
    gemm_f16_wmma<8><<<dim3(BNODE / 128, 1), 256, 0, stream>>>(m2a, m2w2T, m2b2, m2b, BNODE, 128, 128, 0, 2, 0, 0);
    stats_rows_kernel<<<640, 128, 0, stream>>>(m2b, (long)BNODE, statsAll + 768);
    bn_finalize_kernel<<<1, 128, 0, stream>>>(statsAll + 768, (float)BNODE, m2g, m2bb, acAll + 768);

    // ---- node2edge gather + concat, mlp3 + BN ----
    concat_kernel<<<BE, 128, 0, stream>>>(m2b, acAll + 768, xsk, cat);
    gemm_f16_wmma<8><<<dim3(BE / 128, 1), 256, 0, stream>>>(cat, m3w1T, m3b1, m3a, BE, 384, 128, 0, 2, 0, 0);
    gemm_f16_wmma<8><<<dim3(BE / 128, 1), 256, 0, stream>>>(m3a, m3w2T, m3b2, m3b, BE, 128, 128, 0, 2, 0, 0);
    stats_rows_kernel<<<1024, 128, 0, stream>>>(m3b, (long)BE, statsAll + 1024);
    bn_finalize_kernel<<<1, 128, 0, stream>>>(statsAll + 1024, (float)BE, m3g, m3bb, acAll + 1024);

    // ---- fold BN into fc_out, final GEMM -> d_out (f32, [B*E][16]) ----
    fc_fold_kernel<<<1, 32, 0, stream>>>(acAll + 1024, fcw, fcb, fcWT, fcB);
    gemm_f16_wmma<1><<<dim3(BE / 128, 1), 256, 0, stream>>>(m3b, fcWT, fcB, d_out, BE, 128, NOUT, 0, 0, 0, 1);
}
